// tracknet_34179349742393
// MI455X (gfx1250) — matrix operand order repaired and checked by reading, not compiled, not run on hardware
//
#include <hip/hip_runtime.h>

// ---------------------------------------------------------------------------
// CDNA5 (gfx1250) tracking-net forward pass.
// Stride-1 convs: implicit GEMM on V_WMMA_F32_16X16X32_F16 (f16 fragments,
// f32 accumulation). Weight tiles are DMA'd into LDS by the Tensor Data
// Mover (tensor_load_to_lds + s_wait_tensorcnt); im2col tiles staged by the
// workgroup. wave32 / WGP assumptions throughout.
// ---------------------------------------------------------------------------

typedef __attribute__((ext_vector_type(16))) _Float16 v16h;
typedef __attribute__((ext_vector_type(2)))  _Float16 v2h;
typedef __attribute__((ext_vector_type(8)))  float    v8f;
typedef __attribute__((ext_vector_type(4)))  unsigned v4u;
typedef __attribute__((ext_vector_type(8)))  int      v8i;
typedef __attribute__((ext_vector_type(4)))  int      v4i;

#define NB 4          // image batch
#define NF 8          // concat(ims1, ims2) feature batch
#define CORRC 289     // (2*(16/2)+1)^2
#define CFCH 196      // 4 * 7 * 7

#if defined(__gfx1250__) && __has_builtin(__builtin_amdgcn_tensor_load_to_lds)
#define USE_TDM 1
#else
#define USE_TDM 0
#endif

// ----------------------------- conv0 (scalar) ------------------------------
__global__ void k_conv0(const float* __restrict__ im1, const float* __restrict__ im2,
                        const float* __restrict__ w, const float* __restrict__ bias,
                        float* __restrict__ out)
{
    long total = (long)NF * 64 * 256 * 256;
    long idx = (long)blockIdx.x * blockDim.x + threadIdx.x;
    if (idx >= total) return;
    int x  = (int)(idx & 255);
    int y  = (int)((idx >> 8) & 255);
    int co = (int)((idx >> 16) & 63);
    int b  = (int)(idx >> 22);
    const float* src = (b < NB) ? (im1 + (long)b * 3 * 512 * 512)
                                : (im2 + (long)(b - NB) * 3 * 512 * 512);
    float s = bias[co];
    for (int ci = 0; ci < 3; ++ci)
        for (int ky = 0; ky < 3; ++ky) {
            int iy = 2 * y + ky - 1;
            if (iy < 0 || iy >= 512) continue;
            for (int kx = 0; kx < 3; ++kx) {
                int ix = 2 * x + kx - 1;
                if (ix < 0 || ix >= 512) continue;
                s += w[((co * 3 + ci) * 3 + ky) * 3 + kx] *
                     src[((long)ci * 512 + iy) * 512 + ix];
            }
        }
    out[idx] = fmaxf(s, 0.f);
}

// ----------------------------- WMMA conv -----------------------------------
// Block = 128 threads (4 waves): 64 couts x 128 pixels, K in chunks of 32.
// K ordering = (tap major, ci minor); every k>1 conv here has Cin % 32 == 0,
// so a chunk stays within one (ky,kx) tap -> tap decomposition is uniform
// per chunk (scalar). H*W and W are powers of two -> shift/mask pixel math.
__global__ void __launch_bounds__(128)
k_conv_wmma(const float* __restrict__ in, const float* __restrict__ wgt,
            const float* __restrict__ bias, const float* __restrict__ bnS,
            const float* __restrict__ bnB, float* __restrict__ out,
            int N, int Cin, int H, int W, int Cout, int ks, int pad,
            int chOff, int chTot, int doRelu, int hwShift, int wShift)
{
#if USE_TDM
    __shared__ float As32[64][32];    // raw f32 weight tile, filled by TDM
#else
    __shared__ _Float16 As[64][34];   // [m][k] weight tile (manual staging)
#endif
    __shared__ _Float16 Bs[128][34];  // [n][k] im2col tile (k contiguous)

    const int tid  = threadIdx.x;
    const int wave = tid >> 5;
    const int lane = tid & 31;
    const int col  = lane & 15;       // = M for A-frag, = N for B-frag
    const int hi   = lane >> 4;       // upper-half lane selector
    const long p0  = (long)blockIdx.x * 128;
    const int co0  = (int)blockIdx.y * 64;
    const int ks2  = ks * ks;
    const int K    = Cin * ks2;
    const int HWm1 = (1 << hwShift) - 1;
    const int Wm1  = (1 << wShift) - 1;

    v8f acc[8] = {};

    for (int kk = 0; kk < K; kk += 32) {
        if (kk + 32 < K)
            __builtin_prefetch(wgt + (long)(co0 + (tid & 63)) * K + kk + 32, 0, 1);

        // uniform tap decomposition for this chunk (scalar)
        int kykx0 = 0, ci0 = kk;
        if (ks > 1) { kykx0 = kk / Cin; ci0 = kk - kykx0 * Cin; }
        const int ky0 = kykx0 / ks, kx0 = kykx0 - ky0 * ks;

#if USE_TDM
        // ---- A tile via Tensor Data Mover: 1 x 32(ci) x 64(co) f32 tile ----
        // Y stride = ks^2 (ci step), Z stride = K (cout step); per-dim OOB
        // zero-fill handles both the Cin tail (1x1 heads) and the Cout tail.
        if (wave == 0) {
            unsigned lds = (unsigned)(uintptr_t)&As32[0][0];
            unsigned long long ga = (unsigned long long)(uintptr_t)
                (wgt + (long)co0 * K + (long)ci0 * ks2 + kykx0);
            int td1 = Cin - ci0;                 // remaining ci
            int td2 = Cout - co0;                // remaining couts
            v4u g0 = { 1u,                                        // count=1
                       lds,                                       // lds_addr
                       (unsigned)(ga & 0xffffffffu),              // gaddr lo
                       (unsigned)((ga >> 32) & 0x01ffffffu) | (2u << 30) }; // hi|type=2
            v8i g1 = { (int)(2u << 16),                           // data_size=4B
                       (int)(1u << 16),                           // tensor_dim0=1
                       (int)(((unsigned)td1 & 0xffffu) << 16),    // tensor_dim1 lo
                       (int)(((unsigned)td1 >> 16) & 0xffffu) | (1 << 16), // td1 hi | tile_dim0=1
                       (int)((64u << 16) | 32u),                  // tile_dim1=32, tile_dim2=64
                       (int)(unsigned)ks2,                        // dim0_stride lo32
                       (int)(((unsigned)K & 0xffffu) << 16),      // dim0_str hi | dim1_stride lo16
                       (int)((unsigned)K >> 16) };                // dim1_stride hi32
            v4i g2 = { td2, 0, 0, 0 };                            // tensor_dim2
            v4i g3 = { 0, 0, 0, 0 };
#if __clang_major__ >= 23
            v8i g4 = { 0, 0, 0, 0, 0, 0, 0, 0 };
            __builtin_amdgcn_tensor_load_to_lds(g0, g1, g2, g3, g4, 0);
#else
            __builtin_amdgcn_tensor_load_to_lds(g0, g1, g2, g3, 0);
#endif
            __builtin_amdgcn_s_wait_tensorcnt(0);
        }
#else
        // ---- manual A staging fallback ----
        #pragma unroll
        for (int i = 0; i < 16; ++i) {
            int e = i * 128 + tid;
            int m = e >> 5, k = e & 31;
            int co = co0 + m, ci = ci0 + k;
            float v = (co < Cout && ci < Cin)
                      ? wgt[((long)co * Cin + ci) * ks2 + kykx0] : 0.f;
            As[m][k] = (_Float16)v;
        }
#endif
        // ---- B staging: 128 pixels x 32 k (im2col, f32 -> f16) ----
        for (int i = 0; i < 32; ++i) {
            int e = i * 128 + tid;
            int n = e >> 5, k = e & 31;
            int ci = ci0 + k;
            float v = 0.f;
            if (ci < Cin) {
                long pp = p0 + n;
                int b   = (int)(pp >> hwShift);
                int rem = (int)pp & HWm1;
                int y = rem >> wShift, x = rem & Wm1;
                int iy = y + ky0 - pad, ix = x + kx0 - pad;
                if ((unsigned)iy < (unsigned)H && (unsigned)ix < (unsigned)W)
                    v = in[(((long)b * Cin + ci) * H + iy) * W + ix];
            }
            Bs[n][k] = (_Float16)v;
        }
        __syncthreads();

        // ---- A fragment (16x32 f16): lane holds row M=col, K-halves by hi ----
        v16h af;
        {
            const int m = wave * 16 + col;
#if USE_TDM
            #pragma unroll
            for (int v = 0; v < 4; ++v) {
                int k = 2 * v + 8 * hi;
                af[2 * v]     = (_Float16)As32[m][k];
                af[2 * v + 1] = (_Float16)As32[m][k + 1];
            }
            #pragma unroll
            for (int v = 0; v < 4; ++v) {
                int k = 16 + 2 * v + 8 * hi;
                af[8 + 2 * v]     = (_Float16)As32[m][k];
                af[8 + 2 * v + 1] = (_Float16)As32[m][k + 1];
            }
#else
            const v2h* ar = (const v2h*)&As[m][0];
            #pragma unroll
            for (int v = 0; v < 4; ++v) {
                v2h t = ar[v + 4 * hi];            // k = 2v + 8hi
                af[2 * v] = t.x; af[2 * v + 1] = t.y;
            }
            #pragma unroll
            for (int v = 0; v < 4; ++v) {
                v2h t = ar[8 + v + 4 * hi];        // k = 16 + 2v + 8hi
                af[8 + 2 * v] = t.x; af[8 + 2 * v + 1] = t.y;
            }
#endif
        }
        // ---- 8 pixel sub-tiles: B fragment (32x16 f16) + WMMA ----
        #pragma unroll
        for (int s = 0; s < 8; ++s) {
            v16h bf;
            const v2h* br = (const v2h*)&Bs[s * 16 + col][0];
            #pragma unroll
            for (int v = 0; v < 8; ++v) {
                v2h t = br[(v & 3) + 4 * hi + 8 * (v >> 2)]; // k = 2(v&3) + 8hi + 16(v>>2)
                bf[2 * v] = t.x; bf[2 * v + 1] = t.y;
            }
            acc[s] = __builtin_amdgcn_wmma_f32_16x16x32_f16(
                false, af, false, bf, (short)0, acc[s], false, false);
        }
        __syncthreads();
    }

    // ---- epilogue: bias (+ optional BN) (+ optional relu) ----
    #pragma unroll
    for (int s = 0; s < 8; ++s) {
        long pp = p0 + s * 16 + col;
        int b   = (int)(pp >> hwShift);
        int rem = (int)pp & HWm1;
        int y = rem >> wShift, x = rem & Wm1;
        #pragma unroll
        for (int r = 0; r < 8; ++r) {
            int co = co0 + wave * 16 + r + 8 * hi;
            if (co < Cout) {
                float v = acc[s][r] + bias[co];
                if (bnS) v = v * bnS[co] + bnB[co];
                if (doRelu) v = fmaxf(v, 0.f);
                out[(((long)b * chTot + chOff + co) * H + y) * W + x] = v;
            }
        }
    }
}

// ----------------------------- leaf kernels --------------------------------
__global__ void k_maxpool2(const float* __restrict__ in, float* __restrict__ out,
                           int NC, int H, int W)
{
    int Ho = H >> 1, Wo = W >> 1;
    long total = (long)NC * Ho * Wo;
    long idx = (long)blockIdx.x * blockDim.x + threadIdx.x;
    if (idx >= total) return;
    int x = (int)(idx % Wo);
    long t = idx / Wo;
    int y = (int)(t % Ho);
    long nc = t / Ho;
    const float* p = in + (nc * H + 2 * y) * W + 2 * x;
    out[idx] = fmaxf(fmaxf(p[0], p[1]), fmaxf(p[W], p[W + 1]));
}

__global__ void k_resize(const float* __restrict__ in, float* __restrict__ out,
                         int N, int C, int Hin, int Win, int Hout, int Wout,
                         int chOff, int chTot)
{
    long total = (long)N * C * Hout * Wout;
    long idx = (long)blockIdx.x * blockDim.x + threadIdx.x;
    if (idx >= total) return;
    int x = (int)(idx % Wout);
    long t = idx / Wout;
    int y = (int)(t % Hout); t /= Hout;
    int c = (int)(t % C);
    int n = (int)(t / C);
    float fy = (y + 0.5f) * (float)Hin / (float)Hout - 0.5f;
    float fx = (x + 0.5f) * (float)Win / (float)Wout - 0.5f;
    fy = fminf(fmaxf(fy, 0.f), (float)(Hin - 1));
    fx = fminf(fmaxf(fx, 0.f), (float)(Win - 1));
    int y0 = (int)fy, x0 = (int)fx;
    int y1 = min(y0 + 1, Hin - 1), x1 = min(x0 + 1, Win - 1);
    float ay = fy - y0, ax = fx - x0;
    const float* p = in + ((long)n * C + c) * Hin * Win;
    float v = (1.f - ay) * ((1.f - ax) * p[(long)y0 * Win + x0] + ax * p[(long)y0 * Win + x1])
            +        ay  * ((1.f - ax) * p[(long)y1 * Win + x0] + ax * p[(long)y1 * Win + x1]);
    out[(((long)n * chTot + chOff + c) * Hout + y) * Wout + x] = v;
}

__global__ void k_add(const float* __restrict__ a, const float* __restrict__ b,
                      float* __restrict__ o, long n)
{
    long idx = (long)blockIdx.x * blockDim.x + threadIdx.x;
    if (idx < n) o[idx] = a[idx] + b[idx];
}

__global__ void k_corr(const float* __restrict__ f1, const float* __restrict__ f2,
                       float* __restrict__ out, int C, int H, int W)
{
    long total = (long)NB * CORRC * H * W;
    long idx = (long)blockIdx.x * blockDim.x + threadIdx.x;
    if (idx >= total) return;
    int x = (int)(idx % W);
    long t = idx / W;
    int y = (int)(t % H); t /= H;
    int o = (int)(t % CORRC);
    int b = (int)(t / CORRC);
    int iy = y + (o / 17) * 2 - 16;
    int ix = x + (o % 17) * 2 - 16;
    float s = 0.f;
    if (iy >= 0 && iy < H && ix >= 0 && ix < W) {
        long chs = (long)H * W;
        const float* a = f1 + (long)b * C * chs + (long)y * W + x;
        const float* c2 = f2 + (long)b * C * chs + (long)iy * W + ix;
        for (int c = 0; c < C; ++c) s += a[c * chs] * c2[c * chs];
    }
    out[idx] = s / (float)C;
}

__global__ void k_gather_cb(const float* __restrict__ leg2,
                            const float* __restrict__ c1u,
                            const float* __restrict__ c2,
                            float* __restrict__ cb)
{
    long total = (long)NB * 962 * 128 * 128;
    long idx = (long)blockIdx.x * blockDim.x + threadIdx.x;
    if (idx >= total) return;
    long sp = idx & 16383;
    long t = idx >> 14;
    int c = (int)(t % 962);
    int b = (int)(t / 962);
    float v;
    if (c < 192)      v = leg2[(((long)b * 192 + c) << 14) + sp];
    else if (c < 384) v = leg2[((((long)b + 4) * 192 + (c - 192)) << 14) + sp];
    else if (c < 673) v = c1u[(((long)b * CORRC + (c - 384)) << 14) + sp];
    else              v = c2[(((long)b * CORRC + (c - 673)) << 14) + sp];
    cb[idx] = v;
}

__global__ void k_int_row(const float* __restrict__ cf, float* __restrict__ S)
{
    int t = blockIdx.x * blockDim.x + threadIdx.x;
    if (t >= NB * CFCH * 128) return;
    int h = t & 127;
    int bc = t >> 7;
    const float* src = cf + ((long)bc * 128 + h) * 128;
    float* dst = S + (long)bc * 129 * 129 + (long)(h + 1) * 129;
    dst[0] = 0.f;
    float acc = 0.f;
    for (int w = 0; w < 128; ++w) { acc += src[w]; dst[w + 1] = acc; }
}

__global__ void k_int_col(float* __restrict__ S)
{
    int t = blockIdx.x * blockDim.x + threadIdx.x;
    if (t >= NB * CFCH * 129) return;
    int w = t % 129;
    int bc = t / 129;
    float* base = S + (long)bc * 129 * 129;
    base[w] = 0.f;
    float prev = 0.f;
    for (int h = 1; h <= 128; ++h) {
        float cur = base[(long)h * 129 + w] + prev;
        base[(long)h * 129 + w] = cur;
        prev = cur;
    }
}

__global__ void k_psroi(const float* __restrict__ S, const float* __restrict__ rois,
                        float* __restrict__ out)
{
    int t = blockIdx.x * blockDim.x + threadIdx.x;
    if (t >= 128 * 4) return;
    int r = t >> 2, od = t & 3;
    const float* R = rois + r * 5;
    int bi = (int)R[0];
    float x1 = roundf(R[1]) * 0.25f;
    float y1 = roundf(R[2]) * 0.25f;
    float x2 = roundf(R[3] + 1.f) * 0.25f;
    float y2 = roundf(R[4] + 1.f) * 0.25f;
    float bw = fmaxf(x2 - x1, 0.1f) / 7.f;
    float bh = fmaxf(y2 - y1, 0.1f) / 7.f;
    float acc = 0.f;
    for (int gy = 0; gy < 7; ++gy) {
        int hs = (int)fminf(fmaxf(floorf(gy * bh + y1), 0.f), 128.f);
        int he = (int)fminf(fmaxf(ceilf((gy + 1) * bh + y1), 0.f), 128.f);
        for (int gx = 0; gx < 7; ++gx) {
            int wS = (int)fminf(fmaxf(floorf(gx * bw + x1), 0.f), 128.f);
            int wE = (int)fminf(fmaxf(ceilf((gx + 1) * bw + x1), 0.f), 128.f);
            int c = od * 49 + gy * 7 + gx;
            const float* Sp = S + ((long)bi * CFCH + c) * 129 * 129;
            float ssum = Sp[(long)he * 129 + wE] - Sp[(long)hs * 129 + wE]
                       - Sp[(long)he * 129 + wS] + Sp[(long)hs * 129 + wS];
            int area = (he - hs) * (wE - wS);
            acc += (area > 0) ? (ssum / fmaxf((float)area, 1.f)) : 0.f;
        }
    }
    out[t] = acc / 49.f;
}

// ----------------------------- host launcher -------------------------------
enum {
    IN_IMS1 = 0, IN_IMS2, IN_ROIS,
    P_W0, P_B0, P_W1, P_B1, P_W2, P_B2, P_W3, P_B3,
    P_WS0, P_BS0, P_WU1, P_BU1, P_SU1, P_SBU1,
    P_D1K3W, P_D1K3B, P_D1K3S, P_D1K3SB, P_D1K5W, P_D1K5B, P_D1K5S, P_D1K5SB,
    P_D1PW, P_D1PB, P_D1PS, P_D1PSB,
    P_D2K3W, P_D2K3B, P_D2K3S, P_D2K3SB, P_D2K5W, P_D2K5B, P_D2K5S, P_D2K5SB,
    P_D2PW, P_D2PB, P_D2PS, P_D2PSB,
    P_WU2, P_BU2, P_SU2, P_SBU2, P_WC1, P_BC1, P_WC2, P_BC2
};

static const long F1_OFF   = 0L;          // 16,777,216  f1
static const long LEG1_OFF = 16777216L;   // 12,582,912  leg1 / p1 / p2 / x0
static const long F2_OFF   = 29360128L;   //  8,388,608  f2 / corr1 / corr_feat
static const long LEG2_OFF = 37748736L;   // 25,165,824  leg2 / p0 / f3 / u1s
static const long A_OFF    = 62914560L;   // 63,045,632  f0 / d-branches / cb / S

extern "C" void kernel_launch(void* const* d_in, const int* in_sizes, int n_in,
                              void* d_out, int out_size, void* d_ws, size_t ws_size,
                              hipStream_t stream)
{
    (void)in_sizes; (void)n_in; (void)out_size; (void)ws_size;
    float* ws   = (float*)d_ws;
    float* outF = (float*)d_out;
    #define PF(i) ((const float*)d_in[i])

    float* f0    = ws + A_OFF;
    float* p0    = ws + LEG2_OFF;
    float* f1    = ws + F1_OFF;
    float* p1    = ws + LEG1_OFF;
    float* f2    = ws + F2_OFF;
    float* p2    = ws + LEG1_OFF;
    float* f3    = ws + LEG2_OFF;
    float* x0    = ws + LEG1_OFF;
    float* u1s   = ws + LEG2_OFF;
    float* leg1  = ws + LEG1_OFF;
    float* b1d1  = ws + A_OFF;
    float* b2d1  = ws + A_OFF + 4194304L;
    float* u2s   = ws + A_OFF;
    float* leg2  = ws + LEG2_OFF;
    float* b1d2  = ws + A_OFF;
    float* b2d2  = ws + A_OFF + 8388608L;
    float* corr1 = ws + F2_OFF;
    float* corr2 = outF + 512;
    float* c1u   = ws + F1_OFF;
    float* cb    = ws + A_OFF;
    float* cb2   = ws + F1_OFF;
    float* cfeat = ws + F2_OFF;
    float* S     = ws + A_OFF;

    auto conv = [&](const float* in, int wi, int bi, int si, int sbi, float* out_,
                    int N, int Cin, int H, int W, int Cout, int ks, int pad,
                    int chOff, int chTot, int relu) {
        int wsh  = __builtin_ctz((unsigned)W);
        int hwsh = __builtin_ctz((unsigned)(H * W));
        dim3 g((unsigned)(((long)N * H * W) / 128), (unsigned)((Cout + 63) / 64));
        k_conv_wmma<<<g, 128, 0, stream>>>(in, PF(wi), PF(bi),
            si >= 0 ? PF(si) : nullptr, sbi >= 0 ? PF(sbi) : nullptr,
            out_, N, Cin, H, W, Cout, ks, pad, chOff, chTot, relu, hwsh, wsh);
    };
    auto ew = [&](long n) { return dim3((unsigned)((n + 255) / 256)); };

    // backbone
    k_conv0<<<ew((long)NF * 64 * 256 * 256), 256, 0, stream>>>(
        PF(IN_IMS1), PF(IN_IMS2), PF(P_W0), PF(P_B0), f0);
    k_maxpool2<<<ew((long)NF * 64 * 128 * 128), 256, 0, stream>>>(f0, p0, NF * 64, 256, 256);
    conv(p0, P_W1, P_B1, -1, -1, f1, NF, 64, 128, 128, 128, 3, 1, 0, 128, 1);
    k_maxpool2<<<ew((long)NF * 128 * 64 * 64), 256, 0, stream>>>(f1, p1, NF * 128, 128, 128);
    conv(p1, P_W2, P_B2, -1, -1, f2, NF, 128, 64, 64, 256, 3, 1, 0, 256, 1);
    k_maxpool2<<<ew((long)NF * 256 * 32 * 32), 256, 0, stream>>>(f2, p2, NF * 256, 64, 64);
    conv(p2, P_W3, P_B3, -1, -1, f3, NF, 256, 32, 32, 512, 3, 1, 0, 512, 1);
    conv(f3, P_WS0, P_BS0, -1, -1, x0, NF, 512, 32, 32, 256, 3, 1, 0, 256, 1);
    conv(x0, P_WU1, P_BU1, P_SU1, P_SBU1, u1s, NF, 256, 32, 32, 256, 3, 1, 0, 256, 1);
    k_resize<<<ew((long)NF * 256 * 64 * 64), 256, 0, stream>>>(u1s, leg1, NF, 256, 32, 32, 64, 64, 0, 384);

    // dilation_proj d1 on f2 -> leg1[:,256:384]
    conv(f2, P_D1K3W, P_D1K3B, P_D1K3S, P_D1K3SB, b1d1, NF, 256, 64, 64, 128, 3, 1, 0, 128, 1);
    conv(f2, P_D1K5W, P_D1K5B, P_D1K5S, P_D1K5SB, b2d1, NF, 256, 64, 64, 128, 5, 2, 0, 128, 1);
    k_add<<<ew(4194304L), 256, 0, stream>>>(b1d1, b2d1, b1d1, 4194304L);
    conv(b1d1, P_D1PW, P_D1PB, P_D1PS, P_D1PSB, leg1, NF, 128, 64, 64, 128, 1, 0, 256, 384, 1);

    // leg2 = concat(resize(wu2(leg1)), d2(f1))
    conv(leg1, P_WU2, P_BU2, P_SU2, P_SBU2, u2s, NF, 384, 64, 64, 128, 3, 1, 0, 128, 1);
    k_resize<<<ew((long)NF * 128 * 128 * 128), 256, 0, stream>>>(u2s, leg2, NF, 128, 64, 64, 128, 128, 0, 192);
    conv(f1, P_D2K3W, P_D2K3B, P_D2K3S, P_D2K3SB, b1d2, NF, 128, 128, 128, 64, 3, 1, 0, 64, 1);
    conv(f1, P_D2K5W, P_D2K5B, P_D2K5S, P_D2K5SB, b2d2, NF, 128, 128, 128, 64, 5, 2, 0, 64, 1);
    k_add<<<ew(8388608L), 256, 0, stream>>>(b1d2, b2d2, b1d2, 8388608L);
    conv(b1d2, P_D2PW, P_D2PB, P_D2PS, P_D2PSB, leg2, NF, 64, 128, 128, 64, 1, 0, 128, 192, 1);

    // correlations
    k_corr<<<ew((long)NB * CORRC * 64 * 64), 256, 0, stream>>>(
        leg1, leg1 + (long)NB * 384 * 64 * 64, corr1, 384, 64, 64);
    k_corr<<<ew((long)NB * CORRC * 128 * 128), 256, 0, stream>>>(
        leg2, leg2 + (long)NB * 192 * 128 * 128, corr2, 192, 128, 128);
    k_resize<<<ew((long)NB * CORRC * 128 * 128), 256, 0, stream>>>(
        corr1, c1u, NB, CORRC, 64, 64, 128, 128, 0, CORRC);

    // head: cb -> wc1(relu) -> wc2
    k_gather_cb<<<ew((long)NB * 962 * 128 * 128), 256, 0, stream>>>(leg2, c1u, corr2, cb);
    conv(cb,  P_WC1, P_BC1, -1, -1, cb2,   NB, 962, 128, 128, CORRC, 1, 0, 0, CORRC, 1);
    conv(cb2, P_WC2, P_BC2, -1, -1, cfeat, NB, CORRC, 128, 128, CFCH, 1, 0, 0, CFCH, 0);

    // PSROI pooling via integral image, then mean -> d_out[0:512]
    k_int_row<<<ew((long)NB * CFCH * 128), 256, 0, stream>>>(cfeat, S);
    k_int_col<<<ew((long)NB * CFCH * 129), 256, 0, stream>>>(S);
    k_psroi<<<dim3(2), 256, 0, stream>>>(S, PF(IN_ROIS), outF);
    #undef PF
}
